// AttSRU_7834020348299
// MI455X (gfx1250) — matrix operand order repaired and checked by reading, not compiled, not run on hardware
//
#include <hip/hip_runtime.h>

// ---------------------------------------------------------------------------
// Types for CDNA5 WMMA (wave32): v_wmma_f32_16x16x32_bf16
// ---------------------------------------------------------------------------
typedef __attribute__((ext_vector_type(16))) __bf16 v16bf;
typedef __attribute__((ext_vector_type(8)))  __bf16 v8bf;
typedef __attribute__((ext_vector_type(8)))  float  v8f;

#define LN_EPS 1e-6f

// ---------------------------------------------------------------------------
// Elementwise cast (optionally scaled) f32 -> bf16
// ---------------------------------------------------------------------------
__global__ __launch_bounds__(256) void k_cast_scale(const float* __restrict__ X,
                                                    __bf16* __restrict__ Y,
                                                    float scale, long long n) {
  long long i = (long long)blockIdx.x * blockDim.x + threadIdx.x;
  if (i < n) Y[i] = (__bf16)(X[i] * scale);
}

// ---------------------------------------------------------------------------
// Batched transpose + cast: X[R,C] f32 -> Y[C,R] bf16   (R,C multiples of 32)
// ---------------------------------------------------------------------------
__global__ __launch_bounds__(256) void k_transpose_cast(const float* __restrict__ X,
                                                        __bf16* __restrict__ Y,
                                                        int R, int C,
                                                        long long sX, long long sY) {
  __shared__ float tile[32][33];
  X += (long long)blockIdx.z * sX;
  Y += (long long)blockIdx.z * sY;
  int r0 = blockIdx.y * 32, c0 = blockIdx.x * 32;
  int tx = threadIdx.x, ty = threadIdx.y;  // block (32,8)
#pragma unroll
  for (int i = 0; i < 32; i += 8)
    tile[ty + i][tx] = X[(long long)(r0 + ty + i) * C + (c0 + tx)];
  __syncthreads();
#pragma unroll
  for (int i = 0; i < 32; i += 8)
    Y[(long long)(c0 + ty + i) * R + (r0 + tx)] = (__bf16)tile[tx][ty + i];
}

// ---------------------------------------------------------------------------
// Batched WMMA GEMM: C[M,N](f32) = A[M,K](bf16, row-major, lda) x Bt[N,K]^T
// (Bt is N-major so both fragments load contiguously per lane.)
//
// Register-blocked: each wave owns a 2x2 grid of 16x16 f32 accumulators
// (32x32 of C per wave). Block = 256 threads = 8 waves arranged 4(M) x 2(N)
// -> block tile 128x64. K-step 32: 8 b128 loads feed 4 WMMAs per wave.
// Requires M%128==0, N%64==0, K%32==0 (true for all calls here).
// ---------------------------------------------------------------------------
__global__ __launch_bounds__(256) void k_gemm_bt(const __bf16* __restrict__ A,
                                                 long long sA, int lda,
                                                 const __bf16* __restrict__ Bt,
                                                 long long sB, int ldb,
                                                 float* __restrict__ C,
                                                 long long sC, int ldc,
                                                 int M, int N, int K) {
  A  += (long long)blockIdx.z * sA;
  Bt += (long long)blockIdx.z * sB;
  C  += (long long)blockIdx.z * sC;

  const int wave = threadIdx.x >> 5;
  const int lane = threadIdx.x & 31;
  const int wm = wave >> 1;      // 0..3
  const int wn = wave & 1;       // 0..1
  const int m0 = blockIdx.y * 128 + wm * 32;
  const int n0 = blockIdx.x * 64 + wn * 32;
  const int hl = lane >> 4;      // half-wave selector
  const int rc = lane & 15;      // row (A) / col (B)

  // A fragment (16-bit A 16x32): lanes 0-15 hold K={0..7,16..23},
  //                              lanes 16-31 hold K={8..15,24..31}
  const __bf16* ap0 = A + (long long)(m0 + rc) * lda + hl * 8;
  const __bf16* ap1 = ap0 + (long long)16 * lda;
  // B fragment (16-bit B 32x16): lane holds one column; lanes 0-15 K={0..7,16..23},
  //                              lanes 16-31 K={8..15,24..31} (same as A)
  const __bf16* bp0 = Bt + (long long)(n0 + rc) * ldb + hl * 8;
  const __bf16* bp1 = bp0 + (long long)16 * ldb;

  v8f c00 = {}, c01 = {}, c10 = {}, c11 = {};
  for (int k0 = 0; k0 < K; k0 += 32) {
    v16bf a0, a1, b0, b1;
    *((v8bf*)&a0)     = *(const v8bf*)(ap0 + k0);
    *((v8bf*)&a0 + 1) = *(const v8bf*)(ap0 + k0 + 16);
    *((v8bf*)&a1)     = *(const v8bf*)(ap1 + k0);
    *((v8bf*)&a1 + 1) = *(const v8bf*)(ap1 + k0 + 16);
    *((v8bf*)&b0)     = *(const v8bf*)(bp0 + k0);
    *((v8bf*)&b0 + 1) = *(const v8bf*)(bp0 + k0 + 16);
    *((v8bf*)&b1)     = *(const v8bf*)(bp1 + k0);
    *((v8bf*)&b1 + 1) = *(const v8bf*)(bp1 + k0 + 16);
    // prefetch next K-slabs (lowers to global_prefetch_b8)
    __builtin_prefetch(ap0 + k0 + 128, 0, 1);
    __builtin_prefetch(ap1 + k0 + 128, 0, 1);
    __builtin_prefetch(bp0 + k0 + 128, 0, 1);
    __builtin_prefetch(bp1 + k0 + 128, 0, 1);
    c00 = __builtin_amdgcn_wmma_f32_16x16x32_bf16(false, a0, false, b0, (short)0, c00, false, false);
    c01 = __builtin_amdgcn_wmma_f32_16x16x32_bf16(false, a0, false, b1, (short)0, c01, false, false);
    c10 = __builtin_amdgcn_wmma_f32_16x16x32_bf16(false, a1, false, b0, (short)0, c10, false, false);
    c11 = __builtin_amdgcn_wmma_f32_16x16x32_bf16(false, a1, false, b1, (short)0, c11, false, false);
  }

  // C/D layout per 16x16 tile: VGPR r -> row = r + 8*hl ; col = rc
  float* cp00 = C + (long long)(m0 + hl * 8) * ldc + (n0 + rc);
  float* cp01 = cp00 + 16;
  float* cp10 = cp00 + (long long)16 * ldc;
  float* cp11 = cp10 + 16;
#pragma unroll
  for (int r = 0; r < 8; ++r) {
    long long o = (long long)r * ldc;
    cp00[o] = c00[r];
    cp01[o] = c01[r];
    cp10[o] = c10[r];
    cp11[o] = c11[r];
  }
}

// ---------------------------------------------------------------------------
// Row-wise LayerNorm over last dim N; optional f32 (may be in-place) and bf16 outputs
// ---------------------------------------------------------------------------
__global__ __launch_bounds__(256) void k_layernorm(const float* __restrict__ X,
                                                   float* __restrict__ Yf,
                                                   __bf16* __restrict__ Yb,
                                                   const float* __restrict__ g,
                                                   const float* __restrict__ bb,
                                                   int N) {
  long long row = blockIdx.x;
  const float* x = X + row * N;
  __shared__ float s1[256], s2[256];
  float a = 0.f, q = 0.f;
  for (int j = threadIdx.x; j < N; j += 256) { float v = x[j]; a += v; q += v * v; }
  s1[threadIdx.x] = a; s2[threadIdx.x] = q;
  __syncthreads();
  for (int off = 128; off > 0; off >>= 1) {
    if (threadIdx.x < off) {
      s1[threadIdx.x] += s1[threadIdx.x + off];
      s2[threadIdx.x] += s2[threadIdx.x + off];
    }
    __syncthreads();
  }
  float mean = s1[0] / N;
  float var  = s2[0] / N - mean * mean;
  float rs   = rsqrtf(var + LN_EPS);
  float* yf  = Yf ? Yf + row * N : nullptr;
  __bf16* yb = Yb ? Yb + row * N : nullptr;
  for (int j = threadIdx.x; j < N; j += 256) {
    float v = (x[j] - mean) * rs * g[j] + bb[j];
    if (yf) yf[j] = v;
    if (yb) yb[j] = (__bf16)v;
  }
}

// ---------------------------------------------------------------------------
// SRU scan: s_t = (1 - z_t) * s_{t-1} + z_t * p_t  (one thread per (b,d))
// preact is [T,B,3D]; z = sigmoid(col d), p = col 2D+d. Writes ss (bf16) + h_last.
// ---------------------------------------------------------------------------
__global__ __launch_bounds__(256) void k_scan(const float* __restrict__ preact,
                                              const float* __restrict__ h0,
                                              __bf16* __restrict__ ss_bf,
                                              float* __restrict__ h_last,
                                              int T, int Bn, int D) {
  int i = blockIdx.x * blockDim.x + threadIdx.x;
  if (i >= Bn * D) return;
  int b = i / D, d = i % D;
  float s = h0[i];
  const long long rowStride = (long long)Bn * 3 * D;
  const float* pz = preact + (long long)b * 3 * D + d;
  for (int t = 0; t < T; ++t) {
    const float* row = pz + (long long)t * rowStride;
    float z = 1.f / (1.f + __expf(-row[0]));
    float p = row[2 * D];
    s = (1.f - z) * s + z * p;
    ss_bf[((long long)t * Bn + b) * D + d] = (__bf16)s;
  }
  h_last[i] = s;
}

// ---------------------------------------------------------------------------
// Masked, scaled softmax over S. Reads align [B,T,S]; writes p_attn [T,B,S] (f32,
// into d_out) and av [B,T,S] (bf16, for the context GEMM). grid = (T, B).
// ---------------------------------------------------------------------------
__global__ __launch_bounds__(256) void k_softmax(const float* __restrict__ align,
                                                 const int* __restrict__ mlen,
                                                 float* __restrict__ p_attn,
                                                 __bf16* __restrict__ av_bf,
                                                 float scale, int Bn, int T, int S) {
  int t = blockIdx.x, b = blockIdx.y;
  const float* row = align + ((long long)b * T + t) * S;
  int L = mlen[b];
  __shared__ float sm[256];
  float mx = -3.4e38f;
  for (int s = threadIdx.x; s < S; s += 256)
    if (s < L) mx = fmaxf(mx, row[s] * scale);
  sm[threadIdx.x] = mx;
  __syncthreads();
  for (int off = 128; off > 0; off >>= 1) {
    if (threadIdx.x < off) sm[threadIdx.x] = fmaxf(sm[threadIdx.x], sm[threadIdx.x + off]);
    __syncthreads();
  }
  mx = sm[0];
  __syncthreads();
  float sum = 0.f;
  for (int s = threadIdx.x; s < S; s += 256)
    if (s < L) sum += __expf(row[s] * scale - mx);
  sm[threadIdx.x] = sum;
  __syncthreads();
  for (int off = 128; off > 0; off >>= 1) {
    if (threadIdx.x < off) sm[threadIdx.x] += sm[threadIdx.x + off];
    __syncthreads();
  }
  float inv = 1.f / sm[0];
  float* po  = p_attn + ((long long)t * Bn + b) * S;
  __bf16* ao = av_bf + ((long long)b * T + t) * S;
  for (int s = threadIdx.x; s < S; s += 256) {
    float v = (s < L) ? __expf(row[s] * scale - mx) * inv : 0.f;
    po[s] = v;
    ao[s] = (__bf16)v;
  }
}

// ---------------------------------------------------------------------------
// Final: out = (1 - sigmoid(h_gate)) * tanh(t1 + t2) + sigmoid(h_gate) * prev
// ---------------------------------------------------------------------------
__global__ __launch_bounds__(256) void k_final(const float* __restrict__ t1,
                                               const float* __restrict__ t2,
                                               const float* __restrict__ preact,
                                               const float* __restrict__ prev,
                                               float* __restrict__ out,
                                               int D, long long n) {
  long long i = (long long)blockIdx.x * blockDim.x + threadIdx.x;
  if (i >= n) return;
  long long row = i / D;
  int d = (int)(i - row * D);
  float hg = 1.f / (1.f + __expf(-preact[row * 3 * D + D + d]));
  float v = tanhf(t1[i] + t2[i]);
  out[i] = (1.f - hg) * v + hg * prev[i];
}

// ---------------------------------------------------------------------------
// Host orchestration
// ---------------------------------------------------------------------------
extern "C" void kernel_launch(void* const* d_in, const int* in_sizes, int n_in,
                              void* d_out, int out_size, void* d_ws, size_t ws_size,
                              hipStream_t stream) {
  const int T = 512, Bn = 32, S = 512, D = 512;
  const long long TBD = (long long)T * Bn * D;
  const long long BSD = (long long)Bn * S * D;
  const long long BTS = (long long)Bn * T * S;
  const float inv_sqrt_d = 0.04419417382415922f;  // 1/sqrt(512)

  const float* prev   = (const float*)d_in[0];
  const float* hidden = (const float*)d_in[1];
  const float* enc    = (const float*)d_in[2];
  const int*   mlen   = (const int*)d_in[3];
  const float* W_in   = (const float*)d_in[4];
  const float* W_enc  = (const float*)d_in[5];
  const float* W_attn = (const float*)d_in[6];
  const float* W_hid  = (const float*)d_in[7];
  const float* W_ctx  = (const float*)d_in[8];
  const float* g_pre = (const float*)d_in[9],  *b_pre = (const float*)d_in[10];
  const float* g_enc = (const float*)d_in[11], *b_enc = (const float*)d_in[12];
  const float* g_at  = (const float*)d_in[13], *b_at  = (const float*)d_in[14];
  const float* g_h   = (const float*)d_in[15], *b_h   = (const float*)d_in[16];
  const float* g_c   = (const float*)d_in[17], *b_c   = (const float*)d_in[18];

  float* out_main = (float*)d_out;                 // [T,B,D]
  float* h_last   = out_main + TBD;                // [B,D]
  float* p_attn   = h_last + (long long)Bn * D;    // [T,B,S]

  // -- workspace carve (256B aligned) --
  char* ws = (char*)d_ws;
  size_t off = 0;
  auto carve = [&](size_t bytes) -> char* {
    char* p = ws + off;
    off = (off + bytes + 255) & ~(size_t)255;
    return p;
  };
  __bf16* prev_bf    = (__bf16*)carve(TBD * 2);
  __bf16* enc_bf     = (__bf16*)carve(BSD * 2);
  __bf16* Wt_in      = (__bf16*)carve((size_t)3 * D * D * 2);
  __bf16* Wt_enc     = (__bf16*)carve((size_t)D * D * 2);
  __bf16* Wt_attn    = (__bf16*)carve((size_t)D * D * 2);
  __bf16* Wt_hid     = (__bf16*)carve((size_t)D * D * 2);
  __bf16* Wt_ctx     = (__bf16*)carve((size_t)D * D * 2);
  float*  preact     = (float*)carve((size_t)TBD * 3 * 4);  // [T,B,3D]
  float*  pctx       = (float*)carve(BSD * 4);              // [B,S,D]
  __bf16* pctx_bf    = (__bf16*)carve(BSD * 2);
  __bf16* pctxT_bf   = (__bf16*)carve(BSD * 2);             // [B,D,S]
  __bf16* ss_bf      = (__bf16*)carve(TBD * 2);
  float*  tmpA       = (float*)carve(BTS * 4);   // attn_in f32, then align [B,T,S]
  __bf16* attn_in_bf = (__bf16*)carve(TBD * 2);
  __bf16* av_bf      = (__bf16*)carve(BTS * 2);
  float*  cbuf       = (float*)carve(TBD * 4);   // context c, then reused as t1
  __bf16* c_bf       = (__bf16*)carve(TBD * 2);
  float*  t2         = (float*)carve(TBD * 4);
  (void)ws_size; (void)in_sizes; (void)n_in; (void)out_size;

  dim3 blk256(256);

  // 1. casts of activations
  k_cast_scale<<<dim3((TBD + 255) / 256), blk256, 0, stream>>>(prev, prev_bf, 1.f, TBD);
  k_cast_scale<<<dim3((BSD + 255) / 256), blk256, 0, stream>>>(enc, enc_bf, 1.f, BSD);

  // 2. weights -> [N,K] bf16
  k_transpose_cast<<<dim3(3 * D / 32, D / 32, 1), dim3(32, 8), 0, stream>>>(
      W_in, Wt_in, D, 3 * D, 0, 0);
  k_transpose_cast<<<dim3(D / 32, D / 32, 1), dim3(32, 8), 0, stream>>>(
      W_enc, Wt_enc, D, D, 0, 0);
  k_transpose_cast<<<dim3(D / 32, D / 32, 1), dim3(32, 8), 0, stream>>>(
      W_attn, Wt_attn, D, D, 0, 0);
  k_transpose_cast<<<dim3(D / 32, D / 32, 1), dim3(32, 8), 0, stream>>>(
      W_hid, Wt_hid, D, D, 0, 0);
  k_transpose_cast<<<dim3(D / 32, D / 32, 1), dim3(32, 8), 0, stream>>>(
      W_ctx, Wt_ctx, D, D, 0, 0);

  // 3. preact = LN(prev @ W_in)   M=T*B, N=3D, K=D
  k_gemm_bt<<<dim3(3 * D / 64, (T * Bn) / 128, 1), blk256, 0, stream>>>(
      prev_bf, 0, D, Wt_in, 0, D, preact, 0, 3 * D, T * Bn, 3 * D, D);
  k_layernorm<<<dim3(T * Bn), blk256, 0, stream>>>(preact, preact, (__bf16*)nullptr,
                                                   g_pre, b_pre, 3 * D);

  // 4. pctx = LN(enc @ W_enc)     M=B*S, N=D, K=D ; also bf16 copy + per-batch transpose
  k_gemm_bt<<<dim3(D / 64, (Bn * S) / 128, 1), blk256, 0, stream>>>(
      enc_bf, 0, D, Wt_enc, 0, D, pctx, 0, D, Bn * S, D, D);
  k_layernorm<<<dim3(Bn * S), blk256, 0, stream>>>(pctx, pctx, pctx_bf, g_enc, b_enc, D);
  k_transpose_cast<<<dim3(D / 32, S / 32, Bn), dim3(32, 8), 0, stream>>>(
      pctx, pctxT_bf, S, D, (long long)S * D, (long long)S * D);

  // 5. SRU scan -> ss (bf16) and hidden_last (f32, into d_out)
  k_scan<<<dim3((Bn * D) / 256), blk256, 0, stream>>>(preact, hidden, ss_bf, h_last,
                                                      T, Bn, D);

  // 6. attn_in = LN(ss @ W_attn_in) -> bf16
  k_gemm_bt<<<dim3(D / 64, (T * Bn) / 128, 1), blk256, 0, stream>>>(
      ss_bf, 0, D, Wt_attn, 0, D, tmpA, 0, D, T * Bn, D, D);
  k_layernorm<<<dim3(T * Bn), blk256, 0, stream>>>(tmpA, tmpA, attn_in_bf, g_at, b_at, D);

  // 7. align[b,t,s] = sum_d attn_in[t,b,d] * pctx[b,s,d]   (batched over b)
  k_gemm_bt<<<dim3(S / 64, T / 128, Bn), blk256, 0, stream>>>(
      attn_in_bf, (long long)D, Bn * D,          // A: attn_in + b*D, lda=B*D
      pctx_bf, (long long)S * D, D,              // Bt: pctx[b] row-major [S,D]
      tmpA, (long long)T * S, S,                 // C: align[b] [T,S]  (reuses tmpA)
      T, S, D);

  // 8. masked scaled softmax -> p_attn (d_out) + av bf16
  k_softmax<<<dim3(T, Bn), blk256, 0, stream>>>(tmpA, mlen, p_attn, av_bf,
                                                inv_sqrt_d, Bn, T, S);

  // 9. c[t,b,d] = sum_s av[b,t,s] * pctx[b,s,d]   (batched; contraction over S)
  k_gemm_bt<<<dim3(D / 64, T / 128, Bn), blk256, 0, stream>>>(
      av_bf, (long long)T * S, S,                // A: av[b] [T,S]
      pctxT_bf, (long long)D * S, S,             // Bt: pctxT[b] [D,S]
      cbuf, (long long)D, Bn * D,                // C: c + b*D, ldc=B*D -> [T,B,D]
      T, D, S);

  // 10. attn_out = c / sqrt(d) -> bf16 (scale folded into cast)
  k_cast_scale<<<dim3((TBD + 255) / 256), blk256, 0, stream>>>(cbuf, c_bf, inv_sqrt_d, TBD);

  // 11. t1 = LN(ss @ W_hidden)  (reuse cbuf)
  k_gemm_bt<<<dim3(D / 64, (T * Bn) / 128, 1), blk256, 0, stream>>>(
      ss_bf, 0, D, Wt_hid, 0, D, cbuf, 0, D, T * Bn, D, D);
  k_layernorm<<<dim3(T * Bn), blk256, 0, stream>>>(cbuf, cbuf, (__bf16*)nullptr,
                                                   g_h, b_h, D);

  // 12. t2 = LN(attn_out @ W_ctx)
  k_gemm_bt<<<dim3(D / 64, (T * Bn) / 128, 1), blk256, 0, stream>>>(
      c_bf, 0, D, Wt_ctx, 0, D, t2, 0, D, T * Bn, D, D);
  k_layernorm<<<dim3(T * Bn), blk256, 0, stream>>>(t2, t2, (__bf16*)nullptr, g_c, b_c, D);

  // 13. out = (1 - sigmoid(hg)) * tanh(t1 + t2) + sigmoid(hg) * prev
  k_final<<<dim3((TBD + 255) / 256), blk256, 0, stream>>>(cbuf, t2, preact, prev,
                                                          out_main, D, TBD);
}
